// JIIF_conv_mean_33079838113858
// MI455X (gfx1250) — matrix operand order repaired and checked by reading, not compiled, not run on hardware
//
#include <hip/hip_runtime.h>

// ---------------------------------------------------------------------------
// JIIF fusion pipeline on gfx1250 (MI455X): all convs + MLP as WMMA f16 GEMMs.
// Shapes are compile-time (templates) so addressing folds to shifts/adds.
// ---------------------------------------------------------------------------

typedef _Float16 h8   __attribute__((ext_vector_type(8)));
typedef _Float16 v16h __attribute__((ext_vector_type(16)));
typedef float    v8f  __attribute__((ext_vector_type(8)));

union Frag {
    struct { h8 lo; h8 hi; } p;
    v16h v;
};

__device__ __forceinline__ v8f wmma16(const Frag& a, const Frag& b, v8f c) {
    // (neg_a, A, neg_b, B, c_mod, C, reuse_a, reuse_b)
    return __builtin_amdgcn_wmma_f32_16x16x32_f16(false, a.v, false, b.v,
                                                  (short)0, c, false, false);
}

#define BN 2
#define HH 256
#define WWD 256
#define HWPIX (HH * WWD)   // 65536

// ---------------------------------------------------------------------------
// Weight packing: fp32 OIHW -> f16 WMMA-B fragments.
// K ordering: k = (ky*KW+kx)*Cpad + cin  (Cpad multiple of 32).
// Layout: [ntile][kstep][khalf(2)][n16][k16] halves -> per lane a contiguous
// 32B run (two b128 loads), matching dense 16-bit B (32x16) VGPR layout.
// ---------------------------------------------------------------------------
__global__ void pack_convw_kernel(const float* __restrict__ src, _Float16* __restrict__ dst,
                                  int O, int I, int KH, int KW, int Cpad, int CoutPad) {
    int total = CoutPad * KH * KW * Cpad;
    int Ksteps = (KH * KW * Cpad) >> 5;
    for (int idx = blockIdx.x * blockDim.x + threadIdx.x; idx < total;
         idx += gridDim.x * blockDim.x) {
        int cout = idx / (KH * KW * Cpad);
        int rem  = idx % (KH * KW * Cpad);
        int kpos = rem / Cpad;
        int cin  = rem % Cpad;
        float v = 0.f;
        if (cout < O && cin < I) {
            int ky = kpos / KW, kx = kpos % KW;
            v = src[(((size_t)cout * I + cin) * KH + ky) * KW + kx];
        }
        int k = rem;
        size_t o = (((((size_t)(cout >> 4)) * Ksteps + (k >> 5)) * 2 + ((k >> 3) & 1)) * 16
                    + (cout & 15)) * 16 + (((k >> 4) & 1) << 3) + (k & 7);
        dst[o] = (_Float16)v;
    }
}

// MLP weight (K,N) row-major fp32 -> same fragment layout (K padded, N padded)
__global__ void pack_mlpw_kernel(const float* __restrict__ src, _Float16* __restrict__ dst,
                                 int K, int N, int Kpad, int Npad) {
    int total = Npad * Kpad;
    int Ksteps = Kpad >> 5;
    for (int idx = blockIdx.x * blockDim.x + threadIdx.x; idx < total;
         idx += gridDim.x * blockDim.x) {
        int n = idx / Kpad;
        int k = idx % Kpad;
        float v = (n < N && k < K) ? src[(size_t)k * N + n] : 0.f;
        size_t o = (((((size_t)(n >> 4)) * Ksteps + (k >> 5)) * 2 + ((k >> 3) & 1)) * 16
                    + (n & 15)) * 16 + (((k >> 4) & 1) << 3) + (k & 7);
        dst[o] = (_Float16)v;
    }
}

// ---------------------------------------------------------------------------
// Input packing to f16 NHWC with channel padding.
// ---------------------------------------------------------------------------
__global__ void pack_hr_kernel(const float* __restrict__ msi, const float* __restrict__ lms,
                               _Float16* __restrict__ dst) {
    int total = BN * HWPIX * 64;
    for (int idx = blockIdx.x * blockDim.x + threadIdx.x; idx < total;
         idx += gridDim.x * blockDim.x) {
        int b   = idx >> 22;           // / (HWPIX*64)
        int rem = idx & (HWPIX * 64 - 1);
        int pix = rem >> 6;
        int c   = rem & 63;
        float v = 0.f;
        if (c < 3)       v = msi[((size_t)b * 3 + c) * HWPIX + pix];
        else if (c < 34) v = lms[((size_t)b * 31 + (c - 3)) * HWPIX + pix];
        dst[idx] = (_Float16)v;
    }
}

__global__ void pack_lr_kernel(const float* __restrict__ lr, _Float16* __restrict__ dst) {
    int total = BN * 256 * 64;
    for (int idx = blockIdx.x * blockDim.x + threadIdx.x; idx < total;
         idx += gridDim.x * blockDim.x) {
        int b   = idx >> 14;           // / (256*64)
        int rem = idx & (256 * 64 - 1);
        int pix = rem >> 6;
        int c   = rem & 63;
        float v = (c < 34) ? lr[((size_t)b * 34 + c) * 256 + pix] : 0.f;
        dst[idx] = (_Float16)v;
    }
}

// ---------------------------------------------------------------------------
// Generic implicit-GEMM conv (SAME zero pad) with WMMA f16, NHWC f16 tensors.
// Block: 256 threads (8 wave32), tile = 64 pixels x COUT.
// Double-buffered LDS im2col slab, register-staged global loads so the next
// K-step's gather overlaps the current step's WMMAs; one barrier per step.
// ---------------------------------------------------------------------------
template <int CPAD, int COUT, int KH, int KW, int H, int W>
__global__ __launch_bounds__(256) void conv_wmma_kernel(
    const _Float16* __restrict__ in, const _Float16* __restrict__ wpack,
    const float* __restrict__ bias, const _Float16* __restrict__ addbuf,
    _Float16* __restrict__ out, int relu) {
    constexpr int KSTEPS = (KH * KW * CPAD) / 32;
    constexpr int NT     = COUT / 16;
    constexpr int TPW    = (4 * NT) / 8;      // 128ch: 4, 32ch: 1
    constexpr int TPB    = (H * W) / 64;      // pixel tiles per batch (pow2)
    constexpr int PADY   = KH / 2, PADX = KW / 2;

    __shared__ _Float16 Alds[2][64 * 40];     // padded stride 40 halves

    const int tid  = threadIdx.x;
    const int wave = tid >> 5, lane = tid & 31;
    const int b       = blockIdx.x / TPB;     // constant divisor -> shift
    const int pixbase = (blockIdx.x % TPB) * 64;
    const int mtile   = (wave * TPW) / NT;    // constexpr arithmetic
    const int n0      = (wave * TPW) % NT;

    const v8f vz = {0.f, 0.f, 0.f, 0.f, 0.f, 0.f, 0.f, 0.f};
    v8f acc[TPW];
#pragma unroll
    for (int t = 0; t < TPW; ++t) acc[t] = vz;

    // gather mapping: each thread stages 8 contiguous f16 channels of one pixel
    const int gp   = tid >> 2;
    const int gsub = tid & 3;
    const int gpi  = pixbase + gp;
    const int gy0  = gpi / W, gx0 = gpi % W;  // W is pow2 -> shift/mask

    auto loadStage = [&](int ks) -> h8 {
        int kpos = (ks * 32) / CPAD;          // constexpr divisors
        int coff = (ks * 32) % CPAD;
        int y = gy0 + kpos / KW - PADY;
        int x = gx0 + kpos % KW - PADX;
        h8 v = {0, 0, 0, 0, 0, 0, 0, 0};
        if (y >= 0 && y < H && x >= 0 && x < W)
            v = *(const h8*)&in[(((size_t)b * H + y) * W + x) * (size_t)CPAD + coff + gsub * 8];
        return v;
    };

    h8 stage = loadStage(0);
    for (int ks = 0; ks < KSTEPS; ++ks) {
        _Float16* slab = Alds[ks & 1];
        *(h8*)&slab[gp * 40 + gsub * 8] = stage;
        __syncthreads();                       // includes s_wait_dscnt 0
        if (ks + 1 < KSTEPS) stage = loadStage(ks + 1);  // overlaps WMMAs below

        // A fragment: lanes 0-15 -> K {0..7,16..23}, lanes 16-31 -> {8..15,24..31}
        Frag a;
        int m  = mtile * 16 + (lane & 15);
        int k0 = (lane >> 4) << 3;
        a.p.lo = *(const h8*)&slab[m * 40 + k0];
        a.p.hi = *(const h8*)&slab[m * 40 + k0 + 16];
#pragma unroll
        for (int t = 0; t < TPW; ++t) {
            int ntile = n0 + t;
            size_t wb = ((((size_t)ntile * KSTEPS + ks) * 2 + (lane >> 4)) * 16
                         + (lane & 15)) * 16;
            Frag bf;
            bf.p.lo = *(const h8*)&wpack[wb];
            bf.p.hi = *(const h8*)&wpack[wb + 8];
            if (ks + 1 < KSTEPS) __builtin_prefetch(&wpack[wb + 512], 0, 3);
            acc[t] = wmma16(a, bf, acc[t]);
        }
    }

    // C/D layout: lane l, row r -> M = r + 8*(l>=16), N = l&15
#pragma unroll
    for (int t = 0; t < TPW; ++t) {
        int n = (n0 + t) * 16 + (lane & 15);
        float bv = bias ? bias[n] : 0.f;
#pragma unroll
        for (int r = 0; r < 8; ++r) {
            int m = mtile * 16 + r + ((lane >> 4) << 3);
            size_t oidx = ((size_t)b * H * W + (pixbase + m)) * (size_t)COUT + n;
            float v = acc[t][r] + bv;
            if (addbuf) v += (float)addbuf[oidx];
            if (relu) v = v > 0.f ? v : 0.f;
            out[oidx] = (_Float16)v;
        }
    }
}

// ---------------------------------------------------------------------------
// 16x16 max-pool of hr_spa -> feat[...,0:128]
// ---------------------------------------------------------------------------
__global__ void maxpool16_kernel(const _Float16* __restrict__ in, _Float16* __restrict__ feat) {
    int blk = blockIdx.x;            // B*256
    int b = blk >> 8, cell = blk & 255;
    int iy = cell >> 4, ix = cell & 15;
    int c = threadIdx.x;             // 128 threads
    float m = -3.0e38f;
    for (int dy = 0; dy < 16; ++dy)
        for (int dx = 0; dx < 16; ++dx) {
            float v = (float)in[(((size_t)b * HH + iy * 16 + dy) * WWD + ix * 16 + dx) * 128 + c];
            m = v > m ? v : m;
        }
    feat[(((size_t)b * 16 + iy) * 16 + ix) * 256 + c] = (_Float16)m;
}

__global__ void concat_spe_kernel(const _Float16* __restrict__ spe, _Float16* __restrict__ feat) {
    int total = BN * 256 * 128;
    for (int idx = blockIdx.x * blockDim.x + threadIdx.x; idx < total;
         idx += gridDim.x * blockDim.x) {
        int b = idx >> 15;
        int rem = idx & (256 * 128 - 1);
        int cell = rem >> 7, c = rem & 127;
        feat[((size_t)b * 256 + cell) * 256 + 128 + c] = spe[((size_t)b * 256 + cell) * 128 + c];
    }
}

// ---------------------------------------------------------------------------
// Fused implicit MLP: gather -> 386(416)->256->128->33 WMMA GEMMs in LDS,
// 4 shifts, softmax combine. 32 pixels per 256-thread block.
// ---------------------------------------------------------------------------
template <int KSTEPS, int NTILES>
__device__ __forceinline__ void mlp_gemm(const _Float16* X, int xs, _Float16* Y, int ys,
                                         const _Float16* wp, const float* bias,
                                         int wave, int lane) {
    constexpr int TPW = (2 * NTILES) / 8;   // 16 -> 4, 8 -> 2
    const int base  = wave * TPW;
    const int mtile = base / NTILES;        // constant per wave (constexpr divisor)
    const int n0    = base % NTILES;
    const v8f vz = {0.f, 0.f, 0.f, 0.f, 0.f, 0.f, 0.f, 0.f};
    v8f acc[TPW];
#pragma unroll
    for (int t = 0; t < TPW; ++t) acc[t] = vz;

    for (int ks = 0; ks < KSTEPS; ++ks) {
        Frag a;
        int m  = mtile * 16 + (lane & 15);
        int k0 = (ks << 5) + ((lane >> 4) << 3);
        a.p.lo = *(const h8*)&X[m * xs + k0];
        a.p.hi = *(const h8*)&X[m * xs + k0 + 16];
#pragma unroll
        for (int t = 0; t < TPW; ++t) {
            int ntile = n0 + t;
            size_t wb = ((((size_t)ntile * KSTEPS + ks) * 2 + (lane >> 4)) * 16
                         + (lane & 15)) * 16;
            Frag bf;
            bf.p.lo = *(const h8*)&wp[wb];
            bf.p.hi = *(const h8*)&wp[wb + 8];
            acc[t] = wmma16(a, bf, acc[t]);
        }
    }
#pragma unroll
    for (int t = 0; t < TPW; ++t) {
        int n = (n0 + t) * 16 + (lane & 15);
        float bv = bias[n];
#pragma unroll
        for (int r = 0; r < 8; ++r) {
            int m = mtile * 16 + r + ((lane >> 4) << 3);
            float v = acc[t][r] + bv;
            Y[m * ys + n] = (_Float16)(v > 0.f ? v : 0.f);   // relu (layers 1,2)
        }
    }
}

__global__ __launch_bounds__(256) void mlp_fused_kernel(
    const _Float16* __restrict__ feat,    // [B][16][16][256]
    const _Float16* __restrict__ guide,   // [B][256][256][128] (hr_spa)
    const _Float16* __restrict__ w1p, const float* __restrict__ b1,
    const _Float16* __restrict__ w2p, const float* __restrict__ b2,
    const _Float16* __restrict__ w3p, const float* __restrict__ b3,
    _Float16* __restrict__ ret) {         // [B][256][256][32]
    __shared__ __align__(16) unsigned char smem[60928];
    _Float16* Xin = (_Float16*)smem;              // [32][424]; reused as H2 [32][136]
    _Float16* Hb1 = (_Float16*)(smem + 27136);    // [32][264]
    float*    P   = (float*)(smem + 44032);       // [4][32][33]

    const int tid = threadIdx.x, wave = tid >> 5, lane = tid & 31;
    const int b = blockIdx.x >> 11;               // / (HWPIX/32)
    const int pixbase = (blockIdx.x & 2047) * 32;
    const v8f vz = {0.f, 0.f, 0.f, 0.f, 0.f, 0.f, 0.f, 0.f};

    for (int s = 0; s < 4; ++s) {
        float vx = (s < 2) ? -1.f : 1.f;
        float vy = (s & 1) ? 1.f : -1.f;

        // gather q_feat (256 ch, masked nearest sample of 16x16 feat)
        for (int it = 0; it < 4; ++it) {
            int idx = it * 256 + tid;
            int p = idx >> 5, ch = (idx & 31) << 3;
            int pix = pixbase + p, i = pix >> 8, j = pix & 255;
            float cy = (2 * i + 1) * (1.f / 256.f) - 1.f + vx * (1.f / 16.f);
            float cx = (2 * j + 1) * (1.f / 256.f) - 1.f + vy * (1.f / 16.f);
            float fy = rintf(((cy + 1.f) * 16.f - 1.f) * 0.5f);
            float fx = rintf(((cx + 1.f) * 16.f - 1.f) * 0.5f);
            bool valid = (fy >= 0.f) && (fy < 16.f) && (fx >= 0.f) && (fx < 16.f);
            int iy = (int)fminf(fmaxf(fy, 0.f), 15.f);
            int ix = (int)fminf(fmaxf(fx, 0.f), 15.f);
            h8 v = {0, 0, 0, 0, 0, 0, 0, 0};
            if (valid) v = *(const h8*)&feat[(((size_t)b * 16 + iy) * 16 + ix) * 256 + ch];
            *(h8*)&Xin[p * 424 + ch] = v;
        }
        // q_guide_hr: identity sample of hr_spa at the pixel
        for (int it = 0; it < 2; ++it) {
            int idx = it * 256 + tid;
            int p = idx >> 4, ch = (idx & 15) << 3;
            int pix = pixbase + p;
            *(h8*)&Xin[p * 424 + 256 + ch] =
                *(const h8*)&guide[((size_t)b * HWPIX + pix) * 128 + ch];
        }
        // rel coords + zero pad
        if (tid < 32) {
            int p = tid, pix = pixbase + p, i = pix >> 8, j = pix & 255;
            float cy0 = (2 * i + 1) * (1.f / 256.f) - 1.f;
            float cx0 = (2 * j + 1) * (1.f / 256.f) - 1.f;
            float cy = cy0 + vx * (1.f / 16.f), cx = cx0 + vy * (1.f / 16.f);
            float fy = rintf(((cy + 1.f) * 16.f - 1.f) * 0.5f);
            float fx = rintf(((cx + 1.f) * 16.f - 1.f) * 0.5f);
            bool valid = (fy >= 0.f) && (fy < 16.f) && (fx >= 0.f) && (fx < 16.f);
            float qcy = 0.f, qcx = 0.f;
            if (valid) {
                float iy = fminf(fmaxf(fy, 0.f), 15.f);
                float ix = fminf(fmaxf(fx, 0.f), 15.f);
                qcy = -1.f + 1.f / 16.f + iy * (2.f / 16.f);
                qcx = -1.f + 1.f / 16.f + ix * (2.f / 16.f);
            }
            Xin[p * 424 + 384] = (_Float16)((cy0 - qcy) * 16.f);
            Xin[p * 424 + 385] = (_Float16)((cx0 - qcx) * 16.f);
            for (int c = 386; c < 424; ++c) Xin[p * 424 + c] = (_Float16)0.f;
        }
        __syncthreads();

        mlp_gemm<13, 16>(Xin, 424, Hb1, 264, w1p, b1, wave, lane);   // 416->256
        __syncthreads();
        mlp_gemm<8, 8>(Hb1, 264, Xin, 136, w2p, b2, wave, lane);     // 256->128 (H2 aliases Xin)
        __syncthreads();

        // layer 3: 128 -> 48(pad of 33), into preds (wave-uniform predication)
        if (wave < 6) {
            int mtile = wave / 3, ntile = wave % 3;
            v8f acc = vz;
#pragma unroll
            for (int ks = 0; ks < 4; ++ks) {
                Frag a;
                int m  = mtile * 16 + (lane & 15);
                int kk = (ks << 5) + ((lane >> 4) << 3);
                a.p.lo = *(const h8*)&Xin[m * 136 + kk];
                a.p.hi = *(const h8*)&Xin[m * 136 + kk + 16];
                size_t wb = ((((size_t)ntile * 4 + ks) * 2 + (lane >> 4)) * 16
                             + (lane & 15)) * 16;
                Frag bf;
                bf.p.lo = *(const h8*)&w3p[wb];
                bf.p.hi = *(const h8*)&w3p[wb + 8];
                acc = wmma16(a, bf, acc);
            }
            int n = ntile * 16 + (lane & 15);
            if (n < 33) {
                float bv = b3[n];
#pragma unroll
                for (int r = 0; r < 8; ++r) {
                    int m = mtile * 16 + r + ((lane >> 4) << 3);
                    P[((size_t)s * 32 + m) * 33 + n] = acc[r] + bv;
                }
            }
        }
        __syncthreads();
    }

    // softmax over 4 shifts (channel 32) + weighted sum of channels 0..31
    for (int idx = tid; idx < 32 * 32; idx += 256) {
        int p = idx >> 5, c = idx & 31;
        float wv[4];
        float mx = -3.0e38f;
        for (int s = 0; s < 4; ++s) { wv[s] = P[(s * 32 + p) * 33 + 32]; mx = fmaxf(mx, wv[s]); }
        float se = 0.f;
        for (int s = 0; s < 4; ++s) { wv[s] = expf(wv[s] - mx); se += wv[s]; }
        float a = 0.f;
        for (int s = 0; s < 4; ++s) a += P[(s * 32 + p) * 33 + c] * wv[s];
        ret[((size_t)b * HWPIX + pixbase + p) * 32 + c] = (_Float16)(a / se);
    }
}

// out = lms + dec2(dec1(nir))
__global__ void final_add_kernel(const float* __restrict__ lms, const _Float16* __restrict__ x,
                                 float* __restrict__ out, int total) {
    for (int idx = blockIdx.x * blockDim.x + threadIdx.x; idx < total;
         idx += gridDim.x * blockDim.x) {
        int b = idx / (31 * HWPIX);
        int rem = idx % (31 * HWPIX);
        int c = rem >> 16, pix = rem & 65535;
        out[idx] = lms[idx] + (float)x[((size_t)b * HWPIX + pix) * 32 + c];
    }
}

// ---------------------------------------------------------------------------
template <int CPAD, int COUT, int KH, int KW, int H, int W>
static inline void launch_conv(const _Float16* in, const _Float16* wp, const float* bias,
                               const _Float16* add, _Float16* outb, int relu,
                               hipStream_t stream) {
    constexpr int TPB = (H * W) / 64;
    conv_wmma_kernel<CPAD, COUT, KH, KW, H, W><<<BN * TPB, 256, 0, stream>>>(
        in, wp, bias, add, outb, relu);
}

extern "C" void kernel_launch(void* const* d_in, const int* in_sizes, int n_in,
                              void* d_out, int out_size, void* d_ws, size_t ws_size,
                              hipStream_t stream) {
    (void)in_sizes; (void)n_in; (void)out_size; (void)ws_size;
    // Flattened input order (dict/list depth-first):
    const float* HR_MSI = (const float*)d_in[0];
    const float* lms    = (const float*)d_in[1];
    const float* LR_HSI = (const float*)d_in[2];
    // spa: 3 head_w, 4 head_b, 5 tail_w, 6 tail_b, 7..22 blocks (w1,b1,w2,b2)x4
    // spe: 23..42 same layout; mlp: 43..48 (W,b)x3; dec: 49,50
    const float* spaW_raw[10]; const float* spaB[10];
    const float* speW_raw[10]; const float* speB[10];
    spaW_raw[0] = (const float*)d_in[3];  spaB[0] = (const float*)d_in[4];
    spaW_raw[9] = (const float*)d_in[5];  spaB[9] = (const float*)d_in[6];
    speW_raw[0] = (const float*)d_in[23]; speB[0] = (const float*)d_in[24];
    speW_raw[9] = (const float*)d_in[25]; speB[9] = (const float*)d_in[26];
    for (int j = 0; j < 4; ++j) {
        spaW_raw[1 + 2 * j] = (const float*)d_in[7 + 4 * j];
        spaB[1 + 2 * j]     = (const float*)d_in[8 + 4 * j];
        spaW_raw[2 + 2 * j] = (const float*)d_in[9 + 4 * j];
        spaB[2 + 2 * j]     = (const float*)d_in[10 + 4 * j];
        speW_raw[1 + 2 * j] = (const float*)d_in[27 + 4 * j];
        speB[1 + 2 * j]     = (const float*)d_in[28 + 4 * j];
        speW_raw[2 + 2 * j] = (const float*)d_in[29 + 4 * j];
        speB[2 + 2 * j]     = (const float*)d_in[30 + 4 * j];
    }
    const float* mlpW1 = (const float*)d_in[43]; const float* mlpB1 = (const float*)d_in[44];
    const float* mlpW2 = (const float*)d_in[45]; const float* mlpB2 = (const float*)d_in[46];
    const float* mlpW3 = (const float*)d_in[47]; const float* mlpB3 = (const float*)d_in[48];
    const float* decW1_raw = (const float*)d_in[49];
    const float* decW2_raw = (const float*)d_in[50];

    // workspace bump allocator
    char* base = (char*)d_ws;
    size_t off = 0;
    auto allocH = [&](size_t nelem) -> _Float16* {
        _Float16* p = (_Float16*)(base + off);
        off = (off + nelem * sizeof(_Float16) + 255) & ~(size_t)255;
        return p;
    };

    const size_t hr128 = (size_t)BN * HWPIX * 128;
    _Float16* X0 = allocH((size_t)BN * HWPIX * 64);
    _Float16* Xh = allocH(hr128);
    _Float16* Pb = allocH(hr128);
    _Float16* Qb = allocH(hr128);
    _Float16* Rb = allocH(hr128);
    _Float16* L0 = allocH((size_t)BN * 256 * 64);
    _Float16* Lh = allocH((size_t)BN * 256 * 128);
    _Float16* Lp = allocH((size_t)BN * 256 * 128);
    _Float16* Lq = allocH((size_t)BN * 256 * 128);
    _Float16* Lr = allocH((size_t)BN * 256 * 128);
    _Float16* Feat = allocH((size_t)BN * 256 * 256);          // [B][16][16][256]
    _Float16* Xnir = allocH((size_t)BN * HWPIX * 32);
    _Float16* Xd1  = allocH((size_t)BN * HWPIX * 32);
    _Float16* Xd2  = allocH((size_t)BN * HWPIX * 32);

    _Float16* spaW[10]; _Float16* speW[10];
    spaW[0] = allocH((size_t)128 * 9 * 64);                   // head K=576
    speW[0] = allocH((size_t)128 * 9 * 64);
    for (int j = 1; j < 10; ++j) { spaW[j] = allocH((size_t)128 * 9 * 128); }
    for (int j = 1; j < 10; ++j) { speW[j] = allocH((size_t)128 * 9 * 128); }
    _Float16* mW1 = allocH((size_t)256 * 416);
    _Float16* mW2 = allocH((size_t)128 * 256);
    _Float16* mW3 = allocH((size_t)48 * 128);
    _Float16* dW1 = allocH((size_t)32 * 9 * 32);
    _Float16* dW2 = allocH((size_t)32 * 25 * 32);

    auto packConv = [&](const float* src, _Float16* dst, int O, int I, int KH, int KW,
                        int Cpad, int CoutPad) {
        int total = CoutPad * KH * KW * Cpad;
        pack_convw_kernel<<<(total + 255) / 256, 256, 0, stream>>>(src, dst, O, I, KH, KW,
                                                                   Cpad, CoutPad);
    };
    auto packMlp = [&](const float* src, _Float16* dst, int K, int N, int Kpad, int Npad) {
        int total = Npad * Kpad;
        pack_mlpw_kernel<<<(total + 255) / 256, 256, 0, stream>>>(src, dst, K, N, Kpad, Npad);
    };

    // --- weight packs ---
    packConv(spaW_raw[0], spaW[0], 128, 34, 3, 3, 64, 128);
    packConv(speW_raw[0], speW[0], 128, 34, 3, 3, 64, 128);
    for (int j = 1; j < 10; ++j) {
        packConv(spaW_raw[j], spaW[j], 128, 128, 3, 3, 128, 128);
        packConv(speW_raw[j], speW[j], 128, 128, 3, 3, 128, 128);
    }
    packMlp(mlpW1, mW1, 386, 256, 416, 256);
    packMlp(mlpW2, mW2, 256, 128, 256, 128);
    packMlp(mlpW3, mW3, 128, 33, 128, 48);
    packConv(decW1_raw, dW1, 32, 32, 3, 3, 32, 32);
    packConv(decW2_raw, dW2, 31, 32, 5, 5, 32, 32);

    // --- input packing ---
    {
        int t1 = BN * HWPIX * 64;
        pack_hr_kernel<<<(t1 + 255) / 256, 256, 0, stream>>>(HR_MSI, lms, X0);
        int t2 = BN * 256 * 64;
        pack_lr_kernel<<<(t2 + 255) / 256, 256, 0, stream>>>(LR_HSI, L0);
    }

    // --- HR EDSR: x = head; res chain through 4 blocks; spa = tail(res)+x -> Pb ---
    launch_conv< 64, 128, 3, 3, 256, 256>(X0, spaW[0], spaB[0], nullptr, Xh, 0, stream);
    launch_conv<128, 128, 3, 3, 256, 256>(Xh, spaW[1], spaB[1], nullptr, Pb, 1, stream);
    launch_conv<128, 128, 3, 3, 256, 256>(Pb, spaW[2], spaB[2], Xh, Qb, 0, stream);
    launch_conv<128, 128, 3, 3, 256, 256>(Qb, spaW[3], spaB[3], nullptr, Pb, 1, stream);
    launch_conv<128, 128, 3, 3, 256, 256>(Pb, spaW[4], spaB[4], Qb, Rb, 0, stream);
    launch_conv<128, 128, 3, 3, 256, 256>(Rb, spaW[5], spaB[5], nullptr, Pb, 1, stream);
    launch_conv<128, 128, 3, 3, 256, 256>(Pb, spaW[6], spaB[6], Rb, Qb, 0, stream);
    launch_conv<128, 128, 3, 3, 256, 256>(Qb, spaW[7], spaB[7], nullptr, Pb, 1, stream);
    launch_conv<128, 128, 3, 3, 256, 256>(Pb, spaW[8], spaB[8], Qb, Rb, 0, stream);
    launch_conv<128, 128, 3, 3, 256, 256>(Rb, spaW[9], spaB[9], Xh, Pb, 0, stream);  // Pb=hr_spa
    maxpool16_kernel<<<BN * 256, 128, 0, stream>>>(Pb, Feat);

    // --- LR EDSR (16x16) -> lr_spe in Lp ---
    launch_conv< 64, 128, 3, 3, 16, 16>(L0, speW[0], speB[0], nullptr, Lh, 0, stream);
    launch_conv<128, 128, 3, 3, 16, 16>(Lh, speW[1], speB[1], nullptr, Lp, 1, stream);
    launch_conv<128, 128, 3, 3, 16, 16>(Lp, speW[2], speB[2], Lh, Lq, 0, stream);
    launch_conv<128, 128, 3, 3, 16, 16>(Lq, speW[3], speB[3], nullptr, Lp, 1, stream);
    launch_conv<128, 128, 3, 3, 16, 16>(Lp, speW[4], speB[4], Lq, Lr, 0, stream);
    launch_conv<128, 128, 3, 3, 16, 16>(Lr, speW[5], speB[5], nullptr, Lp, 1, stream);
    launch_conv<128, 128, 3, 3, 16, 16>(Lp, speW[6], speB[6], Lr, Lq, 0, stream);
    launch_conv<128, 128, 3, 3, 16, 16>(Lq, speW[7], speB[7], nullptr, Lp, 1, stream);
    launch_conv<128, 128, 3, 3, 16, 16>(Lp, speW[8], speB[8], Lq, Lr, 0, stream);
    launch_conv<128, 128, 3, 3, 16, 16>(Lr, speW[9], speB[9], Lh, Lp, 0, stream);    // Lp=lr_spe
    {
        int t = BN * 256 * 128;
        concat_spe_kernel<<<(t + 255) / 256, 256, 0, stream>>>(Lp, Feat);
    }

    // --- fused implicit MLP (4 shifts + softmax combine) ---
    mlp_fused_kernel<<<BN * (HWPIX / 32), 256, 0, stream>>>(Feat, Pb, mW1, mlpB1,
                                                            mW2, mlpB2, mW3, mlpB3, Xnir);

    // --- decoder convs + residual add ---
    launch_conv<32, 32, 3, 3, 256, 256>(Xnir, dW1, nullptr, nullptr, Xd1, 0, stream);
    launch_conv<32, 32, 5, 5, 256, 256>(Xd1, dW2, nullptr, nullptr, Xd2, 0, stream);
    {
        int total = BN * 31 * HWPIX;
        final_add_kernel<<<(total + 255) / 256, 256, 0, stream>>>(lms, Xd2,
                                                                  (float*)d_out, total);
    }
}
